// SelfLayer_6451040878804
// MI455X (gfx1250) — matrix operand order repaired and checked by reading, not compiled, not run on hardware
//
#include <hip/hip_runtime.h>
#include <hip/hip_bf16.h>

#define B_  4
#define S_  2048
#define E_  1024
#define H_  16
#define KD_ 64

typedef __attribute__((ext_vector_type(16))) _Float16 v16h;
typedef __attribute__((ext_vector_type(8)))  _Float16 v8h;
typedef __attribute__((ext_vector_type(8)))  float    v8f;
typedef __attribute__((ext_vector_type(4)))  int      v4i;

// ---------------------------------------------------------------------------
// gfx1250 async global->LDS copy (GLOBAL_LOAD_ASYNC_TO_LDS_B128, ASYNCcnt).
// Builtin signature (from hipcc diagnostics): (v4i addrspace(1)*,
// v4i addrspace(3)*, imm int offset, imm int cpol).
// ---------------------------------------------------------------------------
#if __has_builtin(__builtin_amdgcn_global_load_async_to_lds_b128)
#define HAVE_ASYNC_LDS 1
typedef __attribute__((address_space(1))) v4i* as1_v4i;
typedef __attribute__((address_space(3))) v4i* as3_v4i;
__device__ __forceinline__ void async_copy16(const void* g, void* l) {
  // 16 bytes per lane, memory -> LDS, no VGPR round-trip.
  // generic->int->AS cast: LDS generic addrs keep the offset in the low bits.
  __builtin_amdgcn_global_load_async_to_lds_b128(
      (as1_v4i)(uintptr_t)g, (as3_v4i)(uintptr_t)l, 0, 0);
}
__device__ __forceinline__ void wait_asynccnt_le2() {
#if __has_builtin(__builtin_amdgcn_s_wait_asynccnt)
  __builtin_amdgcn_s_wait_asynccnt(2);   // must be a constant integer
#else
  asm volatile("s_wait_asynccnt 0x2" ::: "memory");
#endif
}
#else
#define HAVE_ASYNC_LDS 0
#endif

// ---------------------------------------------------------------------------
// Fragment loaders per CDNA5 ISA WMMA VGPR layouts (CDNA5 ISA).
// A-matrix 16x32 f16: lane L holds row M=L%16; halves 0..7 -> K=(L&16?8:0)+i,
// halves 8..15 -> K=16+(L&16?8:0)+i. Row-major storage => two 16B loads.
// ---------------------------------------------------------------------------
__device__ __forceinline__ v16h load_a_frag(const _Float16* base, int ld) {
  const int lane = threadIdx.x & 31;
  const int row  = lane & 15;
  const int k0   = (lane & 16) ? 8 : 0;
  const _Float16* p = base + row * ld + k0;
  v8h lo = *(const v8h*)(p);
  v8h hi = *(const v8h*)(p + 16);
  v16h r;
#pragma unroll
  for (int i = 0; i < 8; ++i) { r[i] = lo[i]; r[i + 8] = hi[i]; }
  return r;
}

// B-matrix 32x16 f16: lane L holds column N=L%16; halves 0..7 -> K=(L&16?8:0)+i, halves 8..15 -> K=16+(L&16?8:0)+i.
// Source stored "N-major with K contiguous" (i.e. B-transposed) => two 16B loads.
__device__ __forceinline__ v16h load_bt_frag(const _Float16* base, int ld) {
  const int lane = threadIdx.x & 31;
  const int n  = lane & 15;
  const int k0 = (lane & 16) ? 8 : 0;
  const _Float16* p = base + n * ld + k0;
  v8h lo = *(const v8h*)(p);
  v8h hi = *(const v8h*)(p + 16);
  v16h r;
#pragma unroll
  for (int i = 0; i < 8; ++i) { r[i] = lo[i]; r[i + 8] = hi[i]; }
  return r;
}

__device__ __forceinline__ v8f wmma_f16(v16h a, v16h b, v8f c) {
  // D = A*B + C, f32 accumulate (v_wmma_f32_16x16x32_f16)
  return __builtin_amdgcn_wmma_f32_16x16x32_f16(false, a, false, b,
                                                (short)0, c, false, false);
}

// ---------------------------------------------------------------------------
// Projection: out[b,h,s,d] (or V transposed: out[b,h,d,s]) = x(b,s,:) . W[h,:,d]
// Block = 128 threads (4 waves); 16(row) x 64(d) tile per head; needs f32->f16
// conversion so staging stays on the sync path.
// ---------------------------------------------------------------------------
__global__ __launch_bounds__(128) void qkv_proj_kernel(
    const float* __restrict__ x,   // (B*S, E) f32
    const float* __restrict__ W,   // (H, E, KD) f32
    _Float16* __restrict__ out,    // (B,H,S,KD) or (B,H,KD,S)
    int transpose_out)
{
  __shared__ _Float16 ldsA[16 * 64];   // x chunk, row-major (ld=64)
  __shared__ _Float16 ldsWt[64 * 64];  // W chunk transposed: [d][e] (ld=64)

  const int tid   = threadIdx.x;
  const int wave  = tid >> 5;
  const int grow0 = blockIdx.x * 16;   // global row (b*S+s)
  const int h     = blockIdx.y;

  v8f acc = {};

  for (int e0 = 0; e0 < E_; e0 += 64) {
    { // stage x tile 16x64 -> f16
      const int m = tid >> 3;
      const int e = (tid & 7) * 8;
#pragma unroll
      for (int half = 0; half < 2; ++half) {
        const float4 f = *(const float4*)(x + (size_t)(grow0 + m) * E_ + e0 + e + half * 4);
        _Float16* dst = ldsA + m * 64 + e + half * 4;
        dst[0] = (_Float16)f.x; dst[1] = (_Float16)f.y;
        dst[2] = (_Float16)f.z; dst[3] = (_Float16)f.w;
      }
    }
    { // stage W chunk 64(e)x64(d), transposed into [d][e]
      const int dq = (tid & 15) * 4;
      const int el = (tid >> 4) * 8;
#pragma unroll
      for (int j = 0; j < 8; ++j) {
        const float4 f = *(const float4*)(W + ((size_t)h * E_ + (e0 + el + j)) * KD_ + dq);
        ldsWt[(dq + 0) * 64 + el + j] = (_Float16)f.x;
        ldsWt[(dq + 1) * 64 + el + j] = (_Float16)f.y;
        ldsWt[(dq + 2) * 64 + el + j] = (_Float16)f.z;
        ldsWt[(dq + 3) * 64 + el + j] = (_Float16)f.w;
      }
    }
    __syncthreads();
    const _Float16* wt = ldsWt + wave * 16 * 64;
    acc = wmma_f16(load_a_frag(ldsA, 64),      load_bt_frag(wt, 64),      acc);
    acc = wmma_f16(load_a_frag(ldsA + 32, 64), load_bt_frag(wt + 32, 64), acc);
    __syncthreads();
  }

  // Epilogue: C-layout lane holds N=lane%16, rows M=r+8*(lane/16)
  const int lane = tid & 31;
  const int n = lane & 15;
  const int g = lane >> 4;
  const int bi = grow0 / S_;
  const int s0 = grow0 % S_;
  const int d  = wave * 16 + n;
  if (!transpose_out) {
    _Float16* o = out + (((size_t)bi * H_ + h) * S_ + s0) * KD_ + d;
#pragma unroll
    for (int r = 0; r < 8; ++r)
      o[(size_t)(r + 8 * g) * KD_] = (_Float16)acc[r];
  } else {
    _Float16* o = out + (((size_t)bi * H_ + h) * KD_ + d) * S_ + s0;
#pragma unroll
    for (int r = 0; r < 8; ++r)
      o[r + 8 * g] = (_Float16)acc[r];
  }
}

// ---------------------------------------------------------------------------
// Flash attention. Block = 256 threads (8 waves); wave owns 16 q rows, full
// d=64 accumulator. 32-wide key tiles double-buffered in LDS via async
// global->LDS copies (ASYNCcnt) so tile i+1 loads overlap tile i's WMMAs.
// ---------------------------------------------------------------------------
__global__ __launch_bounds__(256) void attn_kernel(
    const _Float16* __restrict__ q16,   // (B,H,S,KD)
    const _Float16* __restrict__ k16,   // (B,H,S,KD)
    const _Float16* __restrict__ vt16,  // (B,H,KD,S)
    const float*    __restrict__ mask,  // (B,S,S)
    float*          __restrict__ out)   // (B,S,H*KD)
{
  __shared__ _Float16 ldsK[2][32 * 64];     // K tile, [s][d], double buffered
  __shared__ _Float16 ldsVt[2][64 * 32];    // V^T tile, [d][s], double buffered
  __shared__ _Float16 ldsP[8 * 16 * 32];    // per-wave P scratch, [q][k]

  const int tid  = threadIdx.x;
  const int wave = tid >> 5;
  const int lane = tid & 31;
  const int h = blockIdx.y, b = blockIdx.z;
  const int q0 = blockIdx.x * 128 + wave * 16;
  const size_t bh = (size_t)b * H_ + h;

  const _Float16* qbase  = q16  + (bh * S_ + q0) * KD_;
  const _Float16* kbase  = k16  + bh * S_ * KD_;
  const _Float16* vtbase = vt16 + bh * (size_t)KD_ * S_;
  const float*    mbase  = mask + (size_t)b * S_ * S_;

  // staging geometry (16B per lane for each of K and V^T)
  const int ksl = tid >> 3;            // 0..31 : key row in tile
  const int kdq = (tid & 7) * 8;       // 0..56 : d offset
  const int vdl = tid >> 2;            // 0..63 : d row
  const int vsq = (tid & 3) * 8;       // 0..24 : key offset

#if HAVE_ASYNC_LDS
#define STAGE_TILE(KT, BUF)                                                     \
  do {                                                                          \
    async_copy16(kbase + (size_t)((KT) + ksl) * KD_ + kdq,                      \
                 &ldsK[(BUF)][ksl * 64 + kdq]);                                 \
    async_copy16(vtbase + (size_t)vdl * S_ + (KT) + vsq,                        \
                 &ldsVt[(BUF)][vdl * 32 + vsq]);                                \
  } while (0)
#else
#define STAGE_TILE(KT, BUF)                                                     \
  do {                                                                          \
    *(float4*)(&ldsK[(BUF)][ksl * 64 + kdq]) =                                  \
        *(const float4*)(kbase + (size_t)((KT) + ksl) * KD_ + kdq);             \
    *(float4*)(&ldsVt[(BUF)][vdl * 32 + vsq]) =                                 \
        *(const float4*)(vtbase + (size_t)vdl * S_ + (KT) + vsq);               \
  } while (0)
#endif

  const v16h aQ0 = load_a_frag(qbase, KD_);        // d = 0..31
  const v16h aQ1 = load_a_frag(qbase + 32, KD_);   // d = 32..63

  v8f O[4] = {v8f{}, v8f{}, v8f{}, v8f{}};
  float mrow[8], lrow[8];
#pragma unroll
  for (int r = 0; r < 8; ++r) { mrow[r] = -1e30f; lrow[r] = 0.0f; }

  const int g = lane >> 4;
  const int n = lane & 15;
  _Float16* pw = ldsP + wave * 16 * 32;

  STAGE_TILE(0, 0);                    // prologue: tile 0 in flight

  int buf = 0;
  for (int kt = 0; kt < S_; kt += 32, buf ^= 1) {
    const int ktn = (kt + 32) & (S_ - 1);    // steady-state next tile (wraps)
    __syncthreads();                   // everyone done reading buf^1
    STAGE_TILE(ktn, buf ^ 1);          // overlap next tile with this compute
#if HAVE_ASYNC_LDS
    wait_asynccnt_le2();               // retire tile kt, leave tile ktn in flight
#endif
    __syncthreads();                   // tile kt visible to all waves

    const _Float16* lk = ldsK[buf];
    const _Float16* lv = ldsVt[buf];

    // prefetch next mask row segment (global_prefetch_b8)
    __builtin_prefetch(mbase + (size_t)(q0 + 8 * g) * S_ + ktn + n, 0, 1);

    // scores: two 16(q) x 16(k) sub-tiles, contracting d=64
    float sc[2][8];
#pragma unroll
    for (int j = 0; j < 2; ++j) {
      v8f s = {};
      s = wmma_f16(aQ0, load_bt_frag(lk + j * 16 * 64, 64), s);
      s = wmma_f16(aQ1, load_bt_frag(lk + j * 16 * 64 + 32, 64), s);
#pragma unroll
      for (int r = 0; r < 8; ++r) {
        const int qq = q0 + r + 8 * g;
        const int kk = kt + j * 16 + n;
        const float m = mbase[(size_t)qq * S_ + kk];
        sc[j][r] = (s[r] * 0.125f) * m + (m - 1.0f) * 1e9f;  // /sqrt(64), mask
      }
    }

    // online softmax update (row reductions across the 16-lane N group)
    float scalev[8];
#pragma unroll
    for (int r = 0; r < 8; ++r) {
      float tm = fmaxf(sc[0][r], sc[1][r]);
      tm = fmaxf(tm, __shfl_xor(tm, 1));
      tm = fmaxf(tm, __shfl_xor(tm, 2));
      tm = fmaxf(tm, __shfl_xor(tm, 4));
      tm = fmaxf(tm, __shfl_xor(tm, 8));
      const float newm = fmaxf(mrow[r], tm);
      scalev[r] = __expf(mrow[r] - newm);
      const float p0 = __expf(sc[0][r] - newm);
      const float p1 = __expf(sc[1][r] - newm);
      float rs = p0 + p1;
      rs += __shfl_xor(rs, 1);
      rs += __shfl_xor(rs, 2);
      rs += __shfl_xor(rs, 4);
      rs += __shfl_xor(rs, 8);
      lrow[r] = lrow[r] * scalev[r] + rs;
      mrow[r] = newm;
      // stash P (f16) in per-wave LDS, [q][k] row-major (ld=32)
      pw[(r + 8 * g) * 32 + 0 * 16 + n] = (_Float16)p0;
      pw[(r + 8 * g) * 32 + 1 * 16 + n] = (_Float16)p1;
    }

    // rescale running output, then O += P * V
#pragma unroll
    for (int t = 0; t < 4; ++t)
#pragma unroll
      for (int r = 0; r < 8; ++r) O[t][r] *= scalev[r];

    const v16h aP = load_a_frag(pw, 32);   // 16(q) x 32(k)
#pragma unroll
    for (int t = 0; t < 4; ++t)
      O[t] = wmma_f16(aP, load_bt_frag(lv + t * 16 * 32, 32), O[t]);
  }

  // normalize and write out (B, S, H*KD) f32
#pragma unroll
  for (int r = 0; r < 8; ++r) {
    const float inv = 1.0f / lrow[r];
    const size_t row = ((size_t)b * S_ + q0 + r + 8 * g) * (H_ * KD_) + h * KD_;
#pragma unroll
    for (int t = 0; t < 4; ++t)
      out[row + t * 16 + n] = O[t][r] * inv;
  }
}

extern "C" void kernel_launch(void* const* d_in, const int* in_sizes, int n_in,
                              void* d_out, int out_size, void* d_ws, size_t ws_size,
                              hipStream_t stream) {
  (void)in_sizes; (void)n_in; (void)out_size; (void)ws_size;
  const float* q_in = (const float*)d_in[0];
  const float* k_in = (const float*)d_in[1];
  const float* v_in = (const float*)d_in[2];
  const float* mask = (const float*)d_in[3];
  const float* Wq   = (const float*)d_in[4];
  const float* Wk   = (const float*)d_in[5];
  const float* Wv   = (const float*)d_in[6];
  float* out = (float*)d_out;

  const size_t elems = (size_t)B_ * H_ * S_ * KD_;
  _Float16* q16  = (_Float16*)d_ws;
  _Float16* k16  = q16 + elems;
  _Float16* vt16 = k16 + elems;   // V stored transposed (B,H,KD,S)

  dim3 pgrid(B_ * S_ / 16, H_);
  qkv_proj_kernel<<<pgrid, 128, 0, stream>>>(q_in, Wq, q16, 0);
  qkv_proj_kernel<<<pgrid, 128, 0, stream>>>(k_in, Wk, k16, 0);
  qkv_proj_kernel<<<pgrid, 128, 0, stream>>>(v_in, Wv, vt16, 1);

  dim3 agrid(S_ / 128, H_, B_);
  attn_kernel<<<agrid, 256, 0, stream>>>(q16, k16, vt16, mask, out);
}
